// SimplePoolGCN_34617436406120
// MI455X (gfx1250) — matrix operand order repaired and checked by reading, not compiled, not run on hardware
//
#include <hip/hip_runtime.h>
#include <hip/hip_bf16.h>

// ---------------------------------------------------------------------------
// SimplePoolGCN on MI455X (gfx1250): bf16 WMMA GEMMs + f32 BN/pool epilogues.
// ---------------------------------------------------------------------------

typedef __bf16 bf16;
typedef __attribute__((ext_vector_type(16))) __bf16 v16bf;
typedef __attribute__((ext_vector_type(8)))  __bf16 v8bf;
typedef __attribute__((ext_vector_type(8)))  float  v8f;

#define BNT 128
#define BKT 32
#define TLDS 40    // LDS tile stride (bf16): rows/cols 80B -> 16B-aligned frags, conflict-free

// ---------------------------------------------------------------------------
// Batched GEMM: C[b] = A[b](MxK) * B[b](KxN)  (row-major, strides in elements)
// bf16 operands, f32 accumulate via v_wmma_f32_16x16x32_bf16, OutT in {bf16,f32}
// MW = 16-row M-subtiles per wave (1 or 2); block tile = (64*MW) x 128.
// Requires M%(64*MW)==0, N%128==0, K%32==0 (true for every GEMM in this net).
// grid = (N/128, M/(64*MW), batch), block = 256 (8 waves = 4 Mwaves x 2 Nwaves).
// A staged row-major; B staged TRANSPOSED (K contiguous): every fragment is
// two ds_load_b128. All fragments are hoisted so the MW*4 WMMAs issue
// back-to-back after a single dscnt wait.
// ---------------------------------------------------------------------------
template <typename OutT, int MW>
__global__ __launch_bounds__(256)
void gemm_bf16_wmma(const bf16* __restrict__ A, long sA, int lda,
                    const bf16* __restrict__ B, long sB, int ldb,
                    OutT* __restrict__ C, long sC, int ldc,
                    const float* __restrict__ bias, int K)
{
    constexpr int BMT = 64 * MW;
    __shared__ bf16 As[BMT][TLDS];     // As[row][k]
    __shared__ bf16 Bt[BNT][TLDS];     // Bt[col][k]  (transposed B tile)

    const int tid  = threadIdx.x;
    const int lane = tid & 31;
    const int wid  = tid >> 5;
    const int wm   = wid & 3;        // wave row   (4 waves along M)
    const int wn   = wid >> 2;       // wave col   (2 waves along N)
    const int l15  = lane & 15;
    const bool hi  = lane >= 16;

    const long bofA = (long)blockIdx.z * sA;
    const long bofB = (long)blockIdx.z * sB;
    const long bofC = (long)blockIdx.z * sC;
    const int rowBlk = blockIdx.y * BMT;
    const int colBlk = blockIdx.x * BNT;

    v8f acc[MW][4] = {};

    for (int kk = 0; kk < K; kk += BKT) {
        // ---- stage A rows 0..32*MW-1 (32*MW x 32 bf16 = 512*MW dwords, coalesced) ---------
        #pragma unroll
        for (int i = 0; i < 2 * MW; ++i) {
            int idx = tid + i * 256;
            int r  = idx >> 4;
            int cw = idx & 15;
            *reinterpret_cast<unsigned int*>(&As[r][cw * 2]) =
                *reinterpret_cast<const unsigned int*>(
                    A + bofA + (long)(rowBlk + r) * lda + kk + cw * 2);
        }
        // ---- stage B tile 32x128 with transpose: 2x2 bf16 blocks -------------
        #pragma unroll
        for (int i = 0; i < 4; ++i) {
            int idx = tid + i * 256;        // 0..1023 block ids
            int rp = idx >> 6;              // K-row pair: rows 2rp, 2rp+1
            int cp = idx & 63;              // col pair:   cols 2cp, 2cp+1
            const bf16* gb = B + bofB + (long)(kk + 2 * rp) * ldb + colBlk + 2 * cp;
            unsigned int w0 = *reinterpret_cast<const unsigned int*>(gb);        // {B[2rp][2cp],  B[2rp][2cp+1]}
            unsigned int w1 = *reinterpret_cast<const unsigned int*>(gb + ldb);  // {B[2rp+1][2cp],B[2rp+1][2cp+1]}
            unsigned int lo  = (w0 & 0xffffu) | (w1 << 16);          // {B[2rp][2cp],   B[2rp+1][2cp]}
            unsigned int hi2 = (w0 >> 16) | (w1 & 0xffff0000u);      // {B[2rp][2cp+1], B[2rp+1][2cp+1]}
            *reinterpret_cast<unsigned int*>(&Bt[2 * cp][2 * rp])     = lo;
            *reinterpret_cast<unsigned int*>(&Bt[2 * cp + 1][2 * rp]) = hi2;
        }
        __syncthreads();

        // ---- hoist ALL fragments, then issue WMMAs back-to-back --------------
        // A frag (ISA 16-bit 16x32 layout): lanes 0-15: K=e / 16+e-8,
        //                                   lanes16-31: K=8+e / 24+e-8
        const int kb  = hi ? 8 : 0;
        const int kb2 = hi ? 16 : 0;
        v16bf afrag[MW];
        #pragma unroll
        for (int mi = 0; mi < MW; ++mi) {
            const int row = (wm * MW + mi) * 16 + l15;
            v8bf alo = *reinterpret_cast<const v8bf*>(&As[row][kb]);        // 16B aligned
            v8bf ahi = *reinterpret_cast<const v8bf*>(&As[row][16 + kb]);   // 16B aligned
            #pragma unroll
            for (int e = 0; e < 8; ++e) { afrag[mi][e] = alo[e]; afrag[mi][8 + e] = ahi[e]; }
        }
        v16bf bfrag[4];
        #pragma unroll
        for (int ns = 0; ns < 4; ++ns) {
            const int col = wn * 64 + ns * 16 + l15;
            v8bf blo = *reinterpret_cast<const v8bf*>(&Bt[col][kb]);        // 16B aligned
            v8bf bhi = *reinterpret_cast<const v8bf*>(&Bt[col][16 + kb]);
            #pragma unroll
            for (int e = 0; e < 8; ++e) { bfrag[ns][e] = blo[e]; bfrag[ns][8 + e] = bhi[e]; }
        }
        #pragma unroll
        for (int mi = 0; mi < MW; ++mi)
            #pragma unroll
            for (int ns = 0; ns < 4; ++ns)
                acc[mi][ns] = __builtin_amdgcn_wmma_f32_16x16x32_bf16(
                    false, afrag[mi], false, bfrag[ns], (short)0, acc[mi][ns], false, false);
        __syncthreads();
    }

    // ---- epilogue: C/D layout: vgpr r -> row r + (hi?8:0), col = l15 ---------
    #pragma unroll
    for (int mi = 0; mi < MW; ++mi) {
        #pragma unroll
        for (int ns = 0; ns < 4; ++ns) {
            const int gcol = colBlk + wn * 64 + ns * 16 + l15;
            const float bv = bias ? bias[gcol] : 0.0f;
            #pragma unroll
            for (int r = 0; r < 8; ++r) {
                const int grow = rowBlk + (wm * MW + mi) * 16 + r + (hi ? 8 : 0);
                C[bofC + (long)grow * ldc + gcol] = (OutT)(acc[mi][ns][r] + bv);
            }
        }
    }
}

// ---------------------------------------------------------------------------
__global__ __launch_bounds__(256)
void f32_to_bf16(const float* __restrict__ in, bf16* __restrict__ out, long n)
{
    long i = (long)blockIdx.x * 256 + threadIdx.x;
    if (i < n) out[i] = (bf16)in[i];
}

// Per-channel sum / sum-of-squares over rows (thread-per-channel: coalesced).
// grid = (C/256, rows/256); requires rows % 256 == 0 and C % 256 == 0.
__global__ __launch_bounds__(256)
void bn_stats(const float* __restrict__ h, int C,
              float* __restrict__ sums, float* __restrict__ sumsq)
{
    const int c = blockIdx.x * 256 + threadIdx.x;
    const long r0 = (long)blockIdx.y * 256;
    float s = 0.0f, s2 = 0.0f;
    for (int i = 0; i < 256; ++i) {
        float v = h[(r0 + i) * (long)C + c];
        s += v; s2 += v * v;
    }
    atomicAdd(&sums[c], s);
    atomicAdd(&sumsq[c], s2);
}

__global__ __launch_bounds__(256)
void bn_finalize(const float* __restrict__ sums, const float* __restrict__ sumsq,
                 const float* __restrict__ gamma, const float* __restrict__ beta,
                 float invCount, int C,
                 float* __restrict__ scale, float* __restrict__ shift)
{
    const int c = blockIdx.x * 256 + threadIdx.x;
    if (c >= C) return;
    const float mu   = sums[c] * invCount;
    const float var  = sumsq[c] * invCount - mu * mu;
    const float rstd = rsqrtf(var + 1e-5f);
    const float sc   = gamma[c] * rstd;
    scale[c] = sc;
    shift[c] = beta[c] - mu * sc;
}

// Fused normalize + ReLU + mean-pool over G consecutive nodes; bf16 output
// (feeds the next bf16 GEMM). out is [B, Mout, C]; in is [B, Mout*G, C].
__global__ __launch_bounds__(256)
void bn_relu_pool(const float* __restrict__ h,
                  const float* __restrict__ scale, const float* __restrict__ shift,
                  bf16* __restrict__ out, int C, int Mout, int G, long total)
{
    long idx = (long)blockIdx.x * 256 + threadIdx.x;
    if (idx >= total) return;
    const int  c = (int)(idx % C);
    const long t = idx / C;
    const int  m = (int)(t % Mout);
    const long b = t / Mout;
    const float sc = scale[c], sh = shift[c];
    const long base = ((b * Mout + m) * (long)G) * C + c;
    float acc = 0.0f;
    for (int g = 0; g < G; ++g) {
        float v = h[base + (long)g * C] * sc + sh;
        acc += fmaxf(v, 0.0f);
    }
    out[idx] = (bf16)(acc / (float)G);
}

// ---------------------------------------------------------------------------
// Workspace layout (bytes). Big buffers ping-pong across stages.
// ---------------------------------------------------------------------------
static constexpr size_t OFF_X16   = 0;           // 256*1024*128 bf16 = 64 MB
static constexpr size_t OFF_ADJ0  = 67108864;    // 1024^2 bf16
static constexpr size_t OFF_ADJ1  = 69206016;    // 256^2 bf16
static constexpr size_t OFF_ADJ2  = 69337088;    // 64^2 bf16
static constexpr size_t OFF_W0    = 69345280;    // 128x256 bf16
static constexpr size_t OFF_W1    = 69410816;    // 256x512 bf16
static constexpr size_t OFF_W2    = 69672960;    // 512x1024 bf16
static constexpr size_t OFF_FCW   = 70721536;    // 1024x256 bf16
static constexpr size_t OFF_P1    = 71245824;    // 64 MB bf16: h0 -> h1 -> h2
static constexpr size_t OFF_P2    = 138354688;   // 32 MB bf16: c0d -> c1d -> c2d
static constexpr size_t OFF_P3    = 171909120;   // 256 MB f32: h0b -> h1b -> h2b
static constexpr size_t OFF_STATS = 440344576;   // sums/sumsq/scale/shift (4x4KB)

extern "C" void kernel_launch(void* const* d_in, const int* in_sizes, int n_in,
                              void* d_out, int out_size, void* d_ws, size_t ws_size,
                              hipStream_t stream)
{
    const float* x    = (const float*)d_in[0];
    const float* adj0 = (const float*)d_in[1];
    const float* adj1 = (const float*)d_in[2];
    const float* adj2 = (const float*)d_in[3];
    // d_in[4..6] = nm0/nm1/nm2: consecutive-index groups -> hardcoded strided pool
    const float* W0   = (const float*)d_in[7];   // b0/b1/b2 cancel inside BN
    const float* g0   = (const float*)d_in[9];
    const float* be0  = (const float*)d_in[10];
    const float* W1   = (const float*)d_in[11];
    const float* g1   = (const float*)d_in[13];
    const float* be1  = (const float*)d_in[14];
    const float* W2   = (const float*)d_in[15];
    const float* g2   = (const float*)d_in[17];
    const float* be2  = (const float*)d_in[18];
    const float* fcW  = (const float*)d_in[19];
    const float* fcb  = (const float*)d_in[20];
    float* out = (float*)d_out;

    char* ws = (char*)d_ws;
    bf16* x16   = (bf16*)(ws + OFF_X16);
    bf16* a016  = (bf16*)(ws + OFF_ADJ0);
    bf16* a116  = (bf16*)(ws + OFF_ADJ1);
    bf16* a216  = (bf16*)(ws + OFF_ADJ2);
    bf16* w016  = (bf16*)(ws + OFF_W0);
    bf16* w116  = (bf16*)(ws + OFF_W1);
    bf16* w216  = (bf16*)(ws + OFF_W2);
    bf16* fcw16 = (bf16*)(ws + OFF_FCW);
    bf16* p1    = (bf16*)(ws + OFF_P1);     // h0 / h1 / h2 (bf16, GEMM out)
    bf16* p2    = (bf16*)(ws + OFF_P2);     // c0d / c1d / c2d (bf16, pool out)
    float* p3   = (float*)(ws + OFF_P3);    // h0b / h1b / h2b (f32, BN input)
    float* sums  = (float*)(ws + OFF_STATS);
    float* sumsq = (float*)(ws + OFF_STATS + 4096);
    float* scal  = (float*)(ws + OFF_STATS + 8192);
    float* shif  = (float*)(ws + OFF_STATS + 12288);

    auto cvt = [&](const float* src, bf16* dst, long n) {
        f32_to_bf16<<<dim3((unsigned)((n + 255) / 256)), 256, 0, stream>>>(src, dst, n);
    };

    // ---- downconvert static operands ----
    cvt(x,    x16,   (long)256 * 1024 * 128);
    cvt(adj0, a016,  (long)1024 * 1024);
    cvt(adj1, a116,  (long)256 * 256);
    cvt(adj2, a216,  (long)64 * 64);
    cvt(W0,   w016,  (long)128 * 256);
    cvt(W1,   w116,  (long)256 * 512);
    cvt(W2,   w216,  (long)512 * 1024);
    cvt(fcW,  fcw16, (long)1024 * 256);

    // ================= Block 0 =================
    // h0[b] = adj0(1024x1024) * x[b](1024x128)          -> p1 (bf16)
    gemm_bf16_wmma<bf16, 2><<<dim3(128 / BNT, 1024 / 128, 256), 256, 0, stream>>>(
        a016, 0, 1024, x16, (long)1024 * 128, 128,
        p1, (long)1024 * 128, 128, nullptr, 1024);
    // h0b[b] = h0[b](1024x128) * W0(128x256)            -> p3 (f32)
    gemm_bf16_wmma<float, 2><<<dim3(256 / BNT, 1024 / 128, 256), 256, 0, stream>>>(
        p1, (long)1024 * 128, 128, w016, 0, 256,
        p3, (long)1024 * 256, 256, nullptr, 128);
    // BN stats over 256*1024 rows, C=256
    (void)hipMemsetAsync(ws + OFF_STATS, 0, 8192, stream);
    bn_stats<<<dim3(1, 1024), 256, 0, stream>>>(p3, 256, sums, sumsq);
    bn_finalize<<<dim3(1), 256, 0, stream>>>(sums, sumsq, g0, be0,
                                             1.0f / 262144.0f, 256, scal, shif);
    // normalize + relu + pool4 -> c0d [256,256,256] bf16
    bn_relu_pool<<<dim3(65536), 256, 0, stream>>>(p3, scal, shif, p2,
                                                  256, 256, 4, (long)16777216);

    // ================= Block 1 =================
    // h1[b] = adj1(256x256) * c0d[b](256x256)           -> p1 (bf16)
    gemm_bf16_wmma<bf16, 2><<<dim3(256 / BNT, 256 / 128, 256), 256, 0, stream>>>(
        a116, 0, 256, p2, (long)256 * 256, 256,
        p1, (long)256 * 256, 256, nullptr, 256);
    // h1b[b] = h1[b](256x256) * W1(256x512)             -> p3 (f32)
    gemm_bf16_wmma<float, 2><<<dim3(512 / BNT, 256 / 128, 256), 256, 0, stream>>>(
        p1, (long)256 * 256, 256, w116, 0, 512,
        p3, (long)256 * 512, 512, nullptr, 256);
    (void)hipMemsetAsync(ws + OFF_STATS, 0, 8192, stream);
    bn_stats<<<dim3(2, 256), 256, 0, stream>>>(p3, 512, sums, sumsq);
    bn_finalize<<<dim3(2), 256, 0, stream>>>(sums, sumsq, g1, be1,
                                             1.0f / 65536.0f, 512, scal, shif);
    // pool4 -> c1d [256,64,512] bf16
    bn_relu_pool<<<dim3(32768), 256, 0, stream>>>(p3, scal, shif, p2,
                                                  512, 64, 4, (long)8388608);

    // ================= Block 2 ================= (M=64 -> MW=1 path)
    // h2[b] = adj2(64x64) * c1d[b](64x512)              -> p1 (bf16)
    gemm_bf16_wmma<bf16, 1><<<dim3(512 / BNT, 64 / 64, 256), 256, 0, stream>>>(
        a216, 0, 64, p2, (long)64 * 512, 512,
        p1, (long)64 * 512, 512, nullptr, 64);
    // h2b[b] = h2[b](64x512) * W2(512x1024)             -> p3 (f32)
    gemm_bf16_wmma<float, 1><<<dim3(1024 / BNT, 64 / 64, 256), 256, 0, stream>>>(
        p1, (long)64 * 512, 512, w216, 0, 1024,
        p3, (long)64 * 1024, 1024, nullptr, 512);
    (void)hipMemsetAsync(ws + OFF_STATS, 0, 8192, stream);
    bn_stats<<<dim3(4, 64), 256, 0, stream>>>(p3, 1024, sums, sumsq);
    bn_finalize<<<dim3(4), 256, 0, stream>>>(sums, sumsq, g2, be2,
                                             1.0f / 16384.0f, 1024, scal, shif);
    // pool over all 64 nodes -> c2d [256,1,1024] bf16
    bn_relu_pool<<<dim3(1024), 256, 0, stream>>>(p3, scal, shif, p2,
                                                 1024, 1, 64, (long)262144);

    // ================= FC =================
    // out = c2d(256x1024) * fcW(1024x256) + fcb  (single GEMM, batch=1)
    gemm_bf16_wmma<float, 2><<<dim3(256 / BNT, 256 / 128, 1), 256, 0, stream>>>(
        p2, 0, 1024, fcw16, 0, 256,
        out, 0, 256, fcb, 1024);
}
